// SnareNet_28638841929908
// MI455X (gfx1250) — compile-verified, matrix operand order checked by reading, not run on hardware
//
#include <hip/hip_runtime.h>
#include <hip/hip_bf16.h>

#define BATCH   16384
#define XDIM    256
#define HIDDEN  1024
#define YDIM    256
#define MDIM    256
#define NEWTON_ITERS 10
#define CG_ITERS 10
#define LAMBDA  0.1f
#define BN_EPS  1e-5f
#define TB      16       // batch rows per Newton workgroup

typedef __bf16 bf16;
typedef __attribute__((ext_vector_type(16))) __bf16 v16bf;
typedef __attribute__((ext_vector_type(8)))  float  v8f;

__device__ __forceinline__ v8f wmma_bf16(v16bf a, v16bf b, v8f c) {
  // D = A(16x32) * B(32x16) + C, f32 accumulate
  return __builtin_amdgcn_wmma_f32_16x16x32_bf16(false, a, false, b, (short)0, c,
                                                 false, false);
}

// Async global->LDS copy of 16 bytes (ASYNCcnt-tracked, bypasses VGPRs).
// Low 32 bits of a generic shared pointer are the LDS byte address.
__device__ __forceinline__ void async_cp_b128(void* lds, const void* gptr) {
  unsigned int loff = (unsigned int)(unsigned long long)lds;
  unsigned long long ga = (unsigned long long)gptr;
  asm volatile("global_load_async_to_lds_b128 %0, %1, off"
               :: "v"(loff), "v"(ga) : "memory");
}
__device__ __forceinline__ void async_wait0() {
  asm volatile("s_wait_asynccnt 0" ::: "memory");
}

// ---------------- prep kernels ----------------

__global__ void cvt_kernel(const float* __restrict__ src, bf16* __restrict__ dst, int n) {
  int i = blockIdx.x * blockDim.x + threadIdx.x;
  if (i < n) dst[i] = (bf16)src[i];
}

// dst[C,R] = bf16(src[R,C]^T)
__global__ void tcvt_kernel(const float* __restrict__ src, bf16* __restrict__ dst, int R, int C) {
  int i = blockIdx.x * blockDim.x + threadIdx.x;
  if (i < R * C) {
    int r = i / C, c = i % C;
    dst[(long)c * R + r] = (bf16)src[i];
  }
}

// fold eval-mode BN + layer bias into scale/bias: out = (xW)*s + ((lb-m)*s + b)
__global__ void bnfold_kernel(const float* __restrict__ g, const float* __restrict__ b,
                              const float* __restrict__ m, const float* __restrict__ v,
                              const float* __restrict__ lb,
                              float* __restrict__ scale, float* __restrict__ bias, int n) {
  int i = blockIdx.x * blockDim.x + threadIdx.x;
  if (i < n) {
    float s = g[i] * rsqrtf(v[i] + BN_EPS);
    scale[i] = s;
    bias[i] = (lb[i] - m[i]) * s + b[i];
  }
}

// diag[n] = sum_m A[m,n]^2 + lambda   (one block of 256 threads)
__global__ void diag_kernel(const float* __restrict__ A, float* __restrict__ dg) {
  int n = threadIdx.x;
  float s = 0.f;
  for (int m = 0; m < MDIM; ++m) {
    float a = A[(long)m * YDIM + n];
    s += a * a;
  }
  dg[n] = s + LAMBDA;
}

// ---------------- front-end WMMA GEMM ----------------
// OUT[BATCH, N] = epilogue( X[BATCH,K] @ W[N,K]^T )
// Block tile 128x64, double-buffered async LDS staging of X/W panels.
// 8 waves: wave w owns rows [16w,16w+16) x all 4 column tiles (4 accumulators).
// MODE 0: bf16 out = relu(acc*scale[n] + bias[n])
// MODE 1: f32  out = acc + bias[n]
// MODE 2: f32  out = acc
template <int MODE>
__global__ __launch_bounds__(256) void gemm_wmma(
    const bf16* __restrict__ X, const bf16* __restrict__ W,
    const float* __restrict__ scale, const float* __restrict__ bias,
    bf16* __restrict__ outb, float* __restrict__ outf, int N, int K) {
  __shared__ __align__(64) bf16 XS[2][128][32];   // 16 KB
  __shared__ __align__(64) bf16 WS[2][64][32];    //  8 KB

  const int tid = threadIdx.x;
  const int w = tid >> 5, lane = tid & 31;
  const int lr = lane & 15, lh = lane >> 4;

  const long rowBase = (long)blockIdx.y * 128;
  const long colBase = (long)blockIdx.x * 64;

  // staging: 16B chunks. XS = 512 chunks (2/thread), WS = 256 chunks (1/thread)
  const int xr0 = tid >> 2, xseg = (tid & 3) * 8;  // chunk tid
  const int xr1 = xr0 + 64;                        // chunk tid + 256
  const int wr = tid >> 2, wseg = (tid & 3) * 8;
  const bf16* gx0 = X + (rowBase + xr0) * K + xseg;
  const bf16* gx1 = X + (rowBase + xr1) * K + xseg;
  const bf16* gw  = W + (colBase + wr) * K + wseg;

  auto stage = [&](int buf, int kb) {
    async_cp_b128(&XS[buf][xr0][xseg], gx0 + kb);
    async_cp_b128(&XS[buf][xr1][xseg], gx1 + kb);
    async_cp_b128(&WS[buf][wr][wseg], gw + kb);
  };

  const int srow = w * 16;
  v8f acc0 = {}, acc1 = {}, acc2 = {}, acc3 = {};

  stage(0, 0);
  int buf = 0;
  for (int kb = 0; kb < K; kb += 32) {
    async_wait0();
    __syncthreads();
    if (kb + 32 < K) stage(buf ^ 1, kb + 32);
    // two 16-wide K sub-steps are packed in the lane's v16bf (lh selects half)
    v16bf a  = *(const v16bf*)(&XS[buf][srow + lr][lh * 16]);
    v16bf b0 = *(const v16bf*)(&WS[buf][0 * 16 + lr][lh * 16]);
    v16bf b1 = *(const v16bf*)(&WS[buf][1 * 16 + lr][lh * 16]);
    v16bf b2 = *(const v16bf*)(&WS[buf][2 * 16 + lr][lh * 16]);
    v16bf b3 = *(const v16bf*)(&WS[buf][3 * 16 + lr][lh * 16]);
    acc0 = wmma_bf16(a, b0, acc0);
    acc1 = wmma_bf16(a, b1, acc1);
    acc2 = wmma_bf16(a, b2, acc2);
    acc3 = wmma_bf16(a, b3, acc3);
    __syncthreads();
    buf ^= 1;
  }

  // epilogue: lane holds N=lr of each column tile, M = lh*8+j of this wave's strip
  v8f accs[4] = {acc0, acc1, acc2, acc3};
  for (int ct = 0; ct < 4; ++ct) {
    const long n = colBase + ct * 16 + lr;
    for (int j = 0; j < 8; ++j) {
      const long r = rowBase + srow + lh * 8 + j;
      float v0 = accs[ct][j];
      if (MODE == 0) {
        v0 = fmaxf(v0 * scale[n] + bias[n], 0.f);
        outb[r * N + n] = (bf16)v0;
      } else if (MODE == 1) {
        outf[r * N + n] = v0 + bias[n];
      } else {
        outf[r * N + n] = v0;
      }
    }
  }
}

// ---------------- Newton / CG kernel ----------------
// One WG = 16 batch rows, all state in LDS. Y in/out via Yg (d_out).
__global__ __launch_bounds__(256) void newton_wmma(
    float* __restrict__ Yg, const float* __restrict__ BNDg,
    const bf16* __restrict__ Ab,   // A  [M,Y] bf16 row-major
    const bf16* __restrict__ Atb,  // A^T [Y,M] bf16 row-major
    const float* __restrict__ dg) {
  __shared__ float Y[TB][256];
  __shared__ float BND[TB][256];
  __shared__ float Xv[TB][256];
  __shared__ float Rv[TB][256];
  __shared__ float Pv[TB][256];
  __shared__ float APv[TB][256];
  __shared__ bf16  STG[TB][256];
  __shared__ float red[256];
  __shared__ float rz_s[TB], rzp_s[TB], alpha_s[TB];

  const int tid = threadIdx.x;
  const int w = tid >> 5, lane = tid & 31;
  const int lr = lane & 15, lh = lane >> 4;
  const int r0 = tid >> 4;            // elementwise: row owned by this thread
  const int c0 = (tid & 15) * 16;     // 16 contiguous columns
  const long gbase = (long)blockIdx.x * TB * 256;

  // dst[r][n] = sum_k STG[r][k] * G[n*256+k]  (G = Ab or Atb, L2-resident)
  auto mm = [&](const bf16* __restrict__ G, float (*dst)[256]) {
    const int cc0 = (2 * w) * 16, cc1 = cc0 + 16;
    v8f a0 = {}, a1 = {};
    for (int kb = 0; kb < 256; kb += 32) {
      const int k = kb + lh * 16;
      v16bf a  = *(const v16bf*)(&STG[lr][k]);
      v16bf b0 = *(const v16bf*)(G + (long)(cc0 + lr) * 256 + k);
      v16bf b1 = *(const v16bf*)(G + (long)(cc1 + lr) * 256 + k);
      a0 = wmma_bf16(a, b0, a0);
      a1 = wmma_bf16(a, b1, a1);
    }
    for (int j = 0; j < 8; ++j) {
      const int m = lh * 8 + j;
      dst[m][cc0 + lr] = a0[j];
      dst[m][cc1 + lr] = a1[j];
    }
  };

  for (int i = 0; i < 16; ++i) {
    Y[r0][c0 + i]   = Yg[gbase + r0 * 256 + c0 + i];
    BND[r0][c0 + i] = BNDg[gbase + r0 * 256 + c0 + i];
  }
  __syncthreads();

  for (int it = 0; it < NEWTON_ITERS; ++it) {
    // ---- g_y = Y @ A^T ----
    for (int i = 0; i < 16; ++i) STG[r0][c0 + i] = (bf16)Y[r0][c0 + i];
    __syncthreads();
    mm(Ab, APv);
    __syncthreads();
    // ---- f = -(relu(bl-g) - relu(g-bu)) = relu(g-bu) - relu(bl-g) ----
    for (int i = 0; i < 16; ++i) {
      float g = APv[r0][c0 + i], b = BND[r0][c0 + i];
      float f = fmaxf(g - (b + 1.f), 0.f) - fmaxf((b - 1.f) - g, 0.f);
      STG[r0][c0 + i] = (bf16)f;
    }
    __syncthreads();
    // ---- rhs = f @ A ----
    mm(Atb, Rv);
    __syncthreads();
    // ---- X0 = rhs/diag ----
    for (int i = 0; i < 16; ++i) {
      float x = Rv[r0][c0 + i] / dg[c0 + i];
      Xv[r0][c0 + i] = x;
      STG[r0][c0 + i] = (bf16)x;
    }
    __syncthreads();
    // ---- R0 = rhs - (J^T J + lambda I) X0 ----
    mm(Ab, APv);
    __syncthreads();
    for (int i = 0; i < 16; ++i) STG[r0][c0 + i] = (bf16)APv[r0][c0 + i];
    __syncthreads();
    mm(Atb, APv);
    __syncthreads();
    for (int i = 0; i < 16; ++i)
      Rv[r0][c0 + i] -= APv[r0][c0 + i] + LAMBDA * Xv[r0][c0 + i];
    __syncthreads();

    for (int k = 0; k < CG_ITERS; ++k) {
      // rz = rowsum(R * R / diag)   (Z = R/diag, Jacobi)
      float part = 0.f;
      for (int i = 0; i < 16; ++i) {
        float rr = Rv[r0][c0 + i];
        part += rr * rr / dg[c0 + i];
      }
      red[tid] = part;
      __syncthreads();
      if (tid < TB) {
        float s = 0.f;
        for (int j = 0; j < 16; ++j) s += red[tid * 16 + j];
        rz_s[tid] = s;
      }
      __syncthreads();
      // P = Z + beta*P (P = Z at k==0), stage bf16
      float beta = 0.f;
      if (k > 0) {
        float d = rzp_s[r0];
        beta = rz_s[r0] / (d == 0.f ? 1e-8f : d);
      }
      for (int i = 0; i < 16; ++i) {
        float z = Rv[r0][c0 + i] / dg[c0 + i];
        float p = (k == 0) ? z : z + beta * Pv[r0][c0 + i];
        Pv[r0][c0 + i] = p;
        STG[r0][c0 + i] = (bf16)p;
      }
      __syncthreads();
      // AP = (J^T J + lambda I) P
      mm(Ab, APv);
      __syncthreads();
      for (int i = 0; i < 16; ++i) STG[r0][c0 + i] = (bf16)APv[r0][c0 + i];
      __syncthreads();
      mm(Atb, APv);
      __syncthreads();
      // pAp = rowsum(P * AP) with lambda term folded in
      part = 0.f;
      for (int i = 0; i < 16; ++i) {
        float ap = APv[r0][c0 + i] + LAMBDA * Pv[r0][c0 + i];
        APv[r0][c0 + i] = ap;
        part += Pv[r0][c0 + i] * ap;
      }
      red[tid] = part;
      __syncthreads();
      if (tid < TB) {
        float s = 0.f;
        for (int j = 0; j < 16; ++j) s += red[tid * 16 + j];
        alpha_s[tid] = rz_s[tid] / (s == 0.f ? 1e-8f : s);
        rzp_s[tid] = rz_s[tid];
      }
      __syncthreads();
      const float al = alpha_s[r0];
      for (int i = 0; i < 16; ++i) {
        Xv[r0][c0 + i] += al * Pv[r0][c0 + i];
        Rv[r0][c0 + i] -= al * APv[r0][c0 + i];
      }
      __syncthreads();
    }
    // y -= clip(X, -1, 1)
    for (int i = 0; i < 16; ++i) {
      float u = Xv[r0][c0 + i];
      Y[r0][c0 + i] -= fminf(1.f, fmaxf(-1.f, u));
    }
    __syncthreads();
  }

  for (int i = 0; i < 16; ++i)
    Yg[gbase + r0 * 256 + c0 + i] = Y[r0][c0 + i];
}

// ---------------- host launcher ----------------

extern "C" void kernel_launch(void* const* d_in, const int* in_sizes, int n_in,
                              void* d_out, int out_size, void* d_ws, size_t ws_size,
                              hipStream_t stream) {
  const float* x    = (const float*)d_in[0];
  const float* W1   = (const float*)d_in[1];
  const float* b1   = (const float*)d_in[2];
  const float* bn1g = (const float*)d_in[3];
  const float* bn1b = (const float*)d_in[4];
  const float* bn1m = (const float*)d_in[5];
  const float* bn1v = (const float*)d_in[6];
  const float* W2   = (const float*)d_in[7];
  const float* b2   = (const float*)d_in[8];
  const float* bn2g = (const float*)d_in[9];
  const float* bn2b = (const float*)d_in[10];
  const float* bn2m = (const float*)d_in[11];
  const float* bn2v = (const float*)d_in[12];
  const float* W3   = (const float*)d_in[13];
  const float* b3   = (const float*)d_in[14];
  const float* A    = (const float*)d_in[15];
  const float* Wb   = (const float*)d_in[16];
  float* out = (float*)d_out;
  (void)in_sizes; (void)n_in; (void)out_size; (void)ws_size;

  char* ws = (char*)d_ws;
  size_t off = 0;
  auto alloc = [&](size_t bytes) -> void* {
    void* p = ws + off;
    off += (bytes + 255) & ~(size_t)255;
    return p;
  };

  bf16* xb   = (bf16*)alloc((size_t)BATCH * XDIM * 2);
  bf16* W1b  = (bf16*)alloc((size_t)HIDDEN * XDIM * 2);
  bf16* W2b  = (bf16*)alloc((size_t)HIDDEN * HIDDEN * 2);
  bf16* W3b  = (bf16*)alloc((size_t)YDIM * HIDDEN * 2);
  bf16* Wbt  = (bf16*)alloc((size_t)MDIM * XDIM * 2);
  bf16* Abf  = (bf16*)alloc((size_t)MDIM * YDIM * 2);
  bf16* Atb  = (bf16*)alloc((size_t)YDIM * MDIM * 2);
  bf16* h1b  = (bf16*)alloc((size_t)BATCH * HIDDEN * 2);
  bf16* h2b  = (bf16*)alloc((size_t)BATCH * HIDDEN * 2);
  float* bnd = (float*)alloc((size_t)BATCH * MDIM * 4);
  float* sc1 = (float*)alloc(HIDDEN * 4);
  float* bi1 = (float*)alloc(HIDDEN * 4);
  float* sc2 = (float*)alloc(HIDDEN * 4);
  float* bi2 = (float*)alloc(HIDDEN * 4);
  float* dgb = (float*)alloc(YDIM * 4);

  const int thr = 256;
  cvt_kernel<<<(BATCH * XDIM + thr - 1) / thr, thr, 0, stream>>>(x, xb, BATCH * XDIM);
  cvt_kernel<<<(HIDDEN * XDIM + thr - 1) / thr, thr, 0, stream>>>(W1, W1b, HIDDEN * XDIM);
  cvt_kernel<<<(HIDDEN * HIDDEN + thr - 1) / thr, thr, 0, stream>>>(W2, W2b, HIDDEN * HIDDEN);
  cvt_kernel<<<(YDIM * HIDDEN + thr - 1) / thr, thr, 0, stream>>>(W3, W3b, YDIM * HIDDEN);
  cvt_kernel<<<(MDIM * YDIM + thr - 1) / thr, thr, 0, stream>>>(A, Abf, MDIM * YDIM);
  tcvt_kernel<<<(MDIM * YDIM + thr - 1) / thr, thr, 0, stream>>>(A, Atb, MDIM, YDIM);
  tcvt_kernel<<<(XDIM * MDIM + thr - 1) / thr, thr, 0, stream>>>(Wb, Wbt, XDIM, MDIM);
  bnfold_kernel<<<(HIDDEN + thr - 1) / thr, thr, 0, stream>>>(bn1g, bn1b, bn1m, bn1v, b1, sc1, bi1, HIDDEN);
  bnfold_kernel<<<(HIDDEN + thr - 1) / thr, thr, 0, stream>>>(bn2g, bn2b, bn2m, bn2v, b2, sc2, bi2, HIDDEN);
  diag_kernel<<<1, YDIM, 0, stream>>>(A, dgb);

  // h1 = relu(bn1(x @ W1^T))
  gemm_wmma<0><<<dim3(HIDDEN / 64, BATCH / 128), 256, 0, stream>>>(
      xb, W1b, sc1, bi1, h1b, nullptr, HIDDEN, XDIM);
  // h2 = relu(bn2(h1 @ W2^T))
  gemm_wmma<0><<<dim3(HIDDEN / 64, BATCH / 128), 256, 0, stream>>>(
      h1b, W2b, sc2, bi2, h2b, nullptr, HIDDEN, HIDDEN);
  // y0 = h2 @ W3^T + b3  -> d_out (f32), Newton iterates in place
  gemm_wmma<1><<<dim3(YDIM / 64, BATCH / 128), 256, 0, stream>>>(
      h2b, W3b, nullptr, b3, nullptr, out, YDIM, HIDDEN);
  // bounds = x @ Wb
  gemm_wmma<2><<<dim3(MDIM / 64, BATCH / 128), 256, 0, stream>>>(
      xb, Wbt, nullptr, nullptr, nullptr, bnd, MDIM, XDIM);

  newton_wmma<<<BATCH / TB, 256, 0, stream>>>(out, bnd, Abf, Atb, dgb);
}
